// SequentialEMGPoseLSTM_46986942218557
// MI455X (gfx1250) — matrix operand order repaired and checked by reading, not compiled, not run on hardware
//
#include <hip/hip_runtime.h>

// ---------------------------------------------------------------------------
// 2-layer LSTM (H=256, B=64, T=4096) + LeakyReLU + Linear head, for gfx1250.
// Two persistent workgroups, layer-pipelined:
//   block 0: layer0 for all t  -> h1 ring buffer (global, bf16)
//   block 1: layer1 + head, one step behind, consumes ring
// GEMMs: v_wmma_f32_16x16x32_bf16; weights bf16-packed in B-fragment order
// (L2-resident, re-read per step via laundered AS(1) pointers -> global_load);
// h state ping-pongs in LDS; cell state lives in VGPRs.
// ---------------------------------------------------------------------------

typedef __attribute__((ext_vector_type(16))) __bf16          v16bf;
typedef __attribute__((ext_vector_type(8)))  float           v8f;
typedef __attribute__((ext_vector_type(16))) unsigned short  usv16;
typedef __attribute__((ext_vector_type(8)))  unsigned short  usv8;

#define AS1 __attribute__((address_space(1)))
typedef AS1 unsigned short*        gup;     // global ushort*
typedef const AS1 unsigned short*  gucp;
typedef const AS1 usv16*           gv16cp;
typedef AS1 float*                 gfp;

#define TT     4096
#define BBATCH 64
#define INC    16
#define HIDN   256
#define G4     1024
#define OUTC   20
#define ACOLS  1056        // producer: [h1_e 0..255|h1_o 256..511|x 1024..1055]
                           // consumer: [h1cur 0..255|h2_e 256..511|h2_o 512..767]
#define RING   16

#define PK0_ELEMS (9  * 64 * 512)
#define PK1_ELEMS (16 * 64 * 512)
#define PKH_ELEMS (8  * 2  * 512)
#define PK_TOTAL  (PK0_ELEMS + PK1_ELEMS + PKH_ELEMS)
#define XT_ELEMS  (TT * BBATCH * INC)
#define RING_ELEMS (RING * BBATCH * HIDN)

__device__ __forceinline__ unsigned short f2bf(float f) {
  unsigned int u = __builtin_bit_cast(unsigned int, f);
  u += 0x7FFFu + ((u >> 16) & 1u);
  return (unsigned short)(u >> 16);
}
__device__ __forceinline__ float bf2f(unsigned short s) {
  return __builtin_bit_cast(float, ((unsigned int)s) << 16);
}
__device__ __forceinline__ float sigf(float x)     { return 1.0f / (1.0f + __expf(-x)); }
__device__ __forceinline__ float tanhfast(float x) { return 1.0f - 2.0f / (__expf(2.0f * x) + 1.0f); }

// Launder a global base address in an SGPR pair, then rebuild an explicit
// address_space(1) pointer: defeats LICM (weights re-read from L2 each step,
// never spilled across the t-loop) while keeping global_load selection.
__device__ __forceinline__ gucp opaque_g(const unsigned short* p) {
  unsigned long long v = (unsigned long long)p;
  asm volatile("" : "+s"(v));
  return (gucp)v;
}

// ---------------------------------------------------------------------------
__global__ void pack_weights(const float* __restrict__ Wih0, const float* __restrict__ Whh0,
                             const float* __restrict__ Wih1, const float* __restrict__ Whh1,
                             const float* __restrict__ Whead, unsigned short* __restrict__ pk) {
  int idx = blockIdx.x * 256 + threadIdx.x;
  if (idx >= PK_TOTAL) return;
  float val = 0.0f;
  if (idx < PK0_ELEMS) {
    int tile = idx >> 9, r = idx & 511;
    int d = tile >> 6, nt = tile & 63;
    int lane = r >> 4, e = r & 15;
    int k = ((lane >> 4) << 3) + (e & 7) + ((e >> 3) << 4);
    int n = nt * 16 + (lane & 15);
    if (d < 8) val = Whh0[n * HIDN + d * 32 + k];
    else       val = (k < INC) ? Wih0[n * INC + k] : 0.0f;
  } else if (idx < PK0_ELEMS + PK1_ELEMS) {
    int j = idx - PK0_ELEMS;
    int tile = j >> 9, r = j & 511;
    int d = tile >> 6, nt = tile & 63;
    int lane = r >> 4, e = r & 15;
    int k = ((lane >> 4) << 3) + (e & 7) + ((e >> 3) << 4);
    int n = nt * 16 + (lane & 15);
    val = (d < 8) ? Wih1[n * HIDN + d * 32 + k]
                  : Whh1[n * HIDN + (d - 8) * 32 + k];
  } else {
    int j = idx - PK0_ELEMS - PK1_ELEMS;
    int tile = j >> 9, r = j & 511;
    int d = tile >> 1, nt = tile & 1;
    int lane = r >> 4, e = r & 15;
    int k = ((lane >> 4) << 3) + (e & 7) + ((e >> 3) << 4);
    int n = nt * 16 + (lane & 15);
    val = (n < OUTC) ? Whead[n * HIDN + d * 32 + k] : 0.0f;
  }
  pk[idx] = f2bf(val);
}

__global__ void pack_x(const float* __restrict__ x, unsigned short* __restrict__ xT) {
  int idx = blockIdx.x * 256 + threadIdx.x;
  if (idx >= XT_ELEMS) return;
  int r = idx >> 12;
  int t = idx & (TT - 1);
  xT[(size_t)t * (BBATCH * INC) + r] = f2bf(x[idx]);
}

__global__ void init_flags(int* flags) {
  if (threadIdx.x < 32) flags[threadIdx.x] = 0;
}

// A fragment (16x32 bf16) from LDS, ISA layout.
__device__ __forceinline__ v16bf load_afrag(const unsigned short* Abuf, int mt,
                                            int colbase, int lane) {
  int M = mt * 16 + (lane & 15);
  int base = colbase + ((lane >> 4) << 3);
  const usv8* q = (const usv8*)(Abuf + M * ACOLS + base);
  usv8 lo = q[0];
  usv8 hi = q[2];
  usv16 vv = __builtin_shufflevector(lo, hi, 0,1,2,3,4,5,6,7,8,9,10,11,12,13,14,15);
  return __builtin_bit_cast(v16bf, vv);
}

__device__ __forceinline__ v16bf load_afrag_leaky(const unsigned short* Abuf, int mt,
                                                  int colbase, int lane) {
  int M = mt * 16 + (lane & 15);
  int base = colbase + ((lane >> 4) << 3);
  const unsigned short* p = Abuf + M * ACOLS + base;
  usv16 vv;
#pragma unroll
  for (int e = 0; e < 8; ++e) {
    float a = bf2f(p[e]);      a = (a >= 0.f) ? a : 0.01f * a; vv[e]     = f2bf(a);
    float b = bf2f(p[16 + e]); b = (b >= 0.f) ? b : 0.01f * b; vv[8 + e] = f2bf(b);
  }
  return __builtin_bit_cast(v16bf, vv);
}

__device__ __forceinline__ v16bf load_bfrag_g(gucp tile, int lane) {
  return __builtin_bit_cast(v16bf, ((gv16cp)tile)[lane]);
}

// One output unit (16 batch rows x 16 h-cols, all 4 gates), depth-2 pipelined
// B loads; optional bf16 mirror of h into the global ring (producer only).
__device__ __forceinline__ void lstm_unit(
    unsigned short* Abuf, const unsigned short* pkIn,
    const float* bias, int nK, int baseA, int baseB, int writebase,
    unsigned long long ringSlot, int u, int lane, v8f& creg) {
  gucp pk = opaque_g(pkIn);
  const int mt = u >> 4, jt = u & 15;
  v8f a0 = {}, a1 = {}, a2 = {}, a3 = {};
  gucp tb = pk + (size_t)jt * 512;
  v16bf b0 = load_bfrag_g(tb,            lane);
  v16bf b1 = load_bfrag_g(tb + 16 * 512, lane);
  v16bf b2 = load_bfrag_g(tb + 32 * 512, lane);
  v16bf b3 = load_bfrag_g(tb + 48 * 512, lane);
  for (int d = 0; d < nK; ++d) {
    const int col = (d < 8) ? (baseA + d * 32) : (baseB + (d - 8) * 32);
    const v16bf af = load_afrag(Abuf, mt, col, lane);
    v16bf n0 = b0, n1 = b1, n2 = b2, n3 = b3;
    if (d + 1 < nK) {                    // prefetch next K-step's B fragments
      gucp nb = pk + (size_t)((d + 1) * 64 + jt) * 512;
      n0 = load_bfrag_g(nb,            lane);
      n1 = load_bfrag_g(nb + 16 * 512, lane);
      n2 = load_bfrag_g(nb + 32 * 512, lane);
      n3 = load_bfrag_g(nb + 48 * 512, lane);
    }
    a0 = __builtin_amdgcn_wmma_f32_16x16x32_bf16(false, af, false, b0, (short)0, a0, false, false);
    a1 = __builtin_amdgcn_wmma_f32_16x16x32_bf16(false, af, false, b1, (short)0, a1, false, false);
    a2 = __builtin_amdgcn_wmma_f32_16x16x32_bf16(false, af, false, b2, (short)0, a2, false, false);
    a3 = __builtin_amdgcn_wmma_f32_16x16x32_bf16(false, af, false, b3, (short)0, a3, false, false);
    b0 = n0; b1 = n1; b2 = n2; b3 = n3;
  }
  const int ncol = jt * 16 + (lane & 15);
  const float c0 = bias[ncol],       c1 = bias[256 + ncol];
  const float c2 = bias[512 + ncol], c3 = bias[768 + ncol];
  const int Mb = mt * 16 + ((lane >> 4) << 3);
#pragma unroll
  for (int v = 0; v < 8; ++v) {
    const float ig = sigf(a0[v] + c0);
    const float fg = sigf(a1[v] + c1);
    const float gg = tanhfast(a2[v] + c2);
    const float og = sigf(a3[v] + c3);
    const float cN = fg * creg[v] + ig * gg;
    creg[v] = cN;
    const unsigned short hv = f2bf(og * tanhfast(cN));
    Abuf[(Mb + v) * ACOLS + writebase + ncol] = hv;
    if (ringSlot) ((gup)ringSlot)[(Mb + v) * HIDN + ncol] = hv;
  }
}

__global__ void __launch_bounds__(1024, 1) lstm_persistent(
    const unsigned short* __restrict__ xT,
    const float* __restrict__ bih0, const float* __restrict__ bhh0,
    const float* __restrict__ bih1, const float* __restrict__ bhh1,
    const float* __restrict__ bheadp,
    const unsigned short* __restrict__ pk0,
    const unsigned short* __restrict__ pk1,
    const unsigned short* __restrict__ pkh,
    unsigned short* __restrict__ ring,
    int* __restrict__ flags,
    float* __restrict__ out) {
  __shared__ __attribute__((aligned(64))) unsigned short Abuf[BBATCH * ACOLS];
  __shared__ float biasL[G4];
  __shared__ float bh[32];

  const int  tid  = threadIdx.x;
  const int  lane = tid & 31;
  const int  w    = tid >> 5;
  const bool producer = (blockIdx.x == 0);
  int* const prodF = flags;
  int* const consF = flags + 16;
  const unsigned long long ringBase = (unsigned long long)ring;

  for (int i = tid; i < BBATCH * ACOLS; i += 1024) Abuf[i] = 0;
  biasL[tid] = producer ? (bih0[tid] + bhh0[tid]) : (bih1[tid] + bhh1[tid]);
  if (!producer && tid < 32) bh[tid] = (tid < OUTC) ? bheadp[tid] : 0.0f;
  v8f ca = {}, cb = {};
  __syncthreads();

  if (producer) {
    // ------------------- layer 0 for all t, fills the ring -----------------
    gucp xg = opaque_g(xT);
    for (int t = 0; t < TT; ++t) {
      if (t >= RING) {        // ring-full backpressure (uniform branch)
        if (tid == 0)
          while (__hip_atomic_load(consF, __ATOMIC_ACQUIRE, __HIP_MEMORY_SCOPE_AGENT)
                 < t - RING + 1)
            __builtin_amdgcn_s_sleep(1);
        __syncthreads();
      }
      Abuf[(tid >> 4) * ACOLS + 1024 + (tid & 15)] = xg[(size_t)t * (BBATCH * INC) + tid];
      __syncthreads();
      const int p = t & 1;
      const int h1rd = 256 * p, h1wr = 256 - h1rd;
      const unsigned long long slot =
          ringBase + (unsigned long long)(t & (RING - 1)) * (BBATCH * HIDN * 2);
      lstm_unit(Abuf, pk0, biasL, 9, h1rd, 1024, h1wr, slot, 2 * w,     lane, ca);
      lstm_unit(Abuf, pk0, biasL, 9, h1rd, 1024, h1wr, slot, 2 * w + 1, lane, cb);
      __threadfence();        // make this step's ring stores device-visible
      __syncthreads();
      if (tid == 0)
        __hip_atomic_store(prodF, t + 1, __ATOMIC_RELEASE, __HIP_MEMORY_SCOPE_AGENT);
    }
  } else {
    // ------------------- layer 1 + head, one step behind -------------------
    for (int t = 0; t < TT; ++t) {
      if (tid == 0)
        while (__hip_atomic_load(prodF, __ATOMIC_ACQUIRE, __HIP_MEMORY_SCOPE_AGENT) < t + 1)
          __builtin_amdgcn_s_sleep(1);
      __syncthreads();
      __threadfence();        // acquire: see producer's ring stores
      {                       // ring slot (32KB bf16) -> LDS h1cur (cols 0..255)
        gv16cp src = (gv16cp)(ringBase +
                     (unsigned long long)(t & (RING - 1)) * (BBATCH * HIDN * 2));
        usv16 v = src[tid];
        const int g = tid * 16, M = g >> 8, j = g & 255;
        *(usv16*)(Abuf + M * ACOLS + j) = v;
      }
      __syncthreads();
      const int p = t & 1;
      const int h2rd = 256 + 256 * p, h2wr = 256 + (256 - 256 * p);
      lstm_unit(Abuf, pk1, biasL, 16, 0, h2rd, h2wr, 0ull, 2 * w,     lane, ca);
      lstm_unit(Abuf, pk1, biasL, 16, 0, h2rd, h2wr, 0ull, 2 * w + 1, lane, cb);
      __syncthreads();
      if (w < 8) {            // head: LeakyReLU(h2) x Whead^T
        gucp pkhL = opaque_g(pkh);
        const int mt = w >> 1, nt = w & 1;
        v8f acc = {};
        for (int d = 0; d < 8; ++d) {
          const v16bf af = load_afrag_leaky(Abuf, mt, h2wr + d * 32, lane);
          acc = __builtin_amdgcn_wmma_f32_16x16x32_bf16(
              false, af, false, load_bfrag_g(pkhL + (size_t)(d * 2 + nt) * 512, lane),
              (short)0, acc, false, false);
        }
        const int n = nt * 16 + (lane & 15);
        if (n < OUTC) {
          const float bb = bh[n];
          const int Mb = mt * 16 + ((lane >> 4) << 3);
          gfp og = (gfp)(unsigned long long)out;
#pragma unroll
          for (int v = 0; v < 8; ++v)
            og[(size_t)((Mb + v) * OUTC + n) * TT + t] = acc[v] + bb;
        }
      }
      __syncthreads();
      if (tid == 0)
        __hip_atomic_store(consF, t + 1, __ATOMIC_RELEASE, __HIP_MEMORY_SCOPE_AGENT);
    }
  }
}

extern "C" void kernel_launch(void* const* d_in, const int* in_sizes, int n_in,
                              void* d_out, int out_size, void* d_ws, size_t ws_size,
                              hipStream_t stream) {
  const float* x     = (const float*)d_in[0];
  const float* Wih0  = (const float*)d_in[1];
  const float* Whh0  = (const float*)d_in[2];
  const float* bih0  = (const float*)d_in[3];
  const float* bhh0  = (const float*)d_in[4];
  const float* Wih1  = (const float*)d_in[5];
  const float* Whh1  = (const float*)d_in[6];
  const float* bih1  = (const float*)d_in[7];
  const float* bhh1  = (const float*)d_in[8];
  const float* Whead = (const float*)d_in[9];
  const float* bhead = (const float*)d_in[10];
  (void)in_sizes; (void)n_in; (void)out_size; (void)ws_size;

  unsigned short* pk   = (unsigned short*)d_ws;     // packed bf16 weights (~1.62 MB)
  unsigned short* xT   = pk + PK_TOTAL;             // time-major bf16 x (8 MB)
  unsigned short* ring = xT + XT_ELEMS;             // h1 ring, 16 x 32 KB
  int*            flg  = (int*)(ring + RING_ELEMS); // prod/cons flags

  pack_weights<<<(PK_TOTAL + 255) / 256, 256, 0, stream>>>(Wih0, Whh0, Wih1, Whh1,
                                                           Whead, pk);
  pack_x<<<(XT_ELEMS + 255) / 256, 256, 0, stream>>>(x, xT);
  init_flags<<<1, 64, 0, stream>>>(flg);

  lstm_persistent<<<2, 1024, 0, stream>>>(xT, bih0, bhh0, bih1, bhh1, bhead,
                                          pk,
                                          pk + PK0_ELEMS,
                                          pk + PK0_ELEMS + PK1_ELEMS,
                                          ring, flg, (float*)d_out);
}
